// PointNetCls300_33603824124452
// MI455X (gfx1250) — matrix operand order repaired and checked by reading, not compiled, not run on hardware
//
#include <hip/hip_runtime.h>
#include <hip/hip_bf16.h>
#include <math.h>

// ---------------------------------------------------------------------------
// PointNet + relation-net forward for MI455X (gfx1250, wave32, WMMA bf16)
// Conventions:
//   A operands (weights):      row-major [M,K] bf16
//   B operands (activations):  TRANSPOSED [N,K] bf16 (k-contiguous rows) so
//                              LDS staging is plain 16B chunk copies and is
//                              eligible for GLOBAL_LOAD_ASYNC_TO_LDS_B128.
//   f32 master activations stay channel-major [C, B*N] for BN statistics.
// ---------------------------------------------------------------------------

#define B_     32
#define NPTS   2048
#define BNCOL  (B_ * NPTS)   // 65536
#define DIM    1024
#define KCENT  1024
#define NATTR  100
#define ADIM   300
#define NPAIR  (B_ * NATTR)  // 3200
#define LDT    40            // LDS row stride in ushorts (32 + 8 pad), 80B rows

typedef __bf16 v16bf __attribute__((ext_vector_type(16)));
typedef float  v8f   __attribute__((ext_vector_type(8)));
typedef unsigned int u32x4 __attribute__((ext_vector_type(4)));
typedef int v4i __attribute__((vector_size(16)));   // matches async-LDS builtin param

#if defined(__HIP_DEVICE_COMPILE__)
#  if __has_builtin(__builtin_amdgcn_global_load_async_to_lds_b128)
#    define HAVE_ASYNC 1
#  else
#    define HAVE_ASYNC 0
#  endif
#else
#  define HAVE_ASYNC 0
#endif

#define AS_GLOBAL __attribute__((address_space(1)))
#define AS_LDS    __attribute__((address_space(3)))

__device__ __forceinline__ unsigned short f2bf(float f) {
  unsigned int x = __float_as_uint(f);
  x += 0x7FFFu + ((x >> 16) & 1u);          // round-to-nearest-even
  return (unsigned short)(x >> 16);
}

// 16-byte global->LDS chunk copy (async on CDNA5 when available)
__device__ __forceinline__ void cp16(const unsigned short* g, unsigned short* l) {
#if HAVE_ASYNC
  __builtin_amdgcn_global_load_async_to_lds_b128(
      (AS_GLOBAL v4i*)g, (AS_LDS v4i*)l, 0, 0);
#else
  *(u32x4*)l = *(const u32x4*)g;
#endif
}

// ---------------------------------------------------------------------------
// Generic WMMA GEMM: C[M,N] = A[M,K]bf16 * Bt[N,K]bf16 (+bias[M])
// block = 256 threads = 8 waves; block tile 128x128, K-step 32.
// Per-chunk hybrid staging: aligned in-range 16B chunks go through the async
// LDS path; edge chunks use guarded scalar zero-fill.
// ---------------------------------------------------------------------------
__global__ __launch_bounds__(256)
void k_gemm(const unsigned short* __restrict__ A,
            const unsigned short* __restrict__ Bt,
            float* __restrict__ C,
            const float* __restrict__ bias,
            int M, int N, int K)
{
  __shared__ __align__(16) unsigned short As[128 * LDT];  // [m][k]
  __shared__ __align__(16) unsigned short Bs[128 * LDT];  // [n][k]
  const int tid    = threadIdx.x;
  const int lane   = tid & 31, wid = tid >> 5;
  const int warpM  = (wid & 3) * 32;
  const int warpN  = (wid >> 2) * 64;
  const int laneLo = lane & 15, laneHf = lane >> 4;
  const int rowBase = blockIdx.y * 128;
  const int colBase = blockIdx.x * 128;
  const bool alignedK = ((K & 7) == 0);

  v8f acc[2][4];
  for (int tr = 0; tr < 2; tr++)
    for (int tc = 0; tc < 4; tc++)
      for (int i = 0; i < 8; i++) acc[tr][tc][i] = 0.f;

  for (int k0 = 0; k0 < K; k0 += 32) {
    if (k0 + 32 < K) {
      int pr = rowBase + (tid >> 1);
      if (pr < M) __builtin_prefetch(&A[(long)pr * K + k0 + 32], 0, 0);
    }
    // stage A tile: 128 rows x 32 k, 16B chunks
    for (int q = tid; q < 512; q += 256) {
      int r = q >> 2, c = (q & 3) * 8;
      int gr = rowBase + r, gk = k0 + c;
      if (alignedK && gr < M && gk + 8 <= K) {
        cp16(&A[(long)gr * K + gk], &As[r * LDT + c]);
      } else {
#pragma unroll
        for (int j = 0; j < 8; j++)
          As[r * LDT + c + j] = (gr < M && gk + j < K) ? A[(long)gr * K + gk + j]
                                                       : (unsigned short)0;
      }
    }
    // stage B tile (Bt is [N,K], k-contiguous)
    for (int q = tid; q < 512; q += 256) {
      int n = q >> 2, c = (q & 3) * 8;
      int gn = colBase + n, gk = k0 + c;
      if (alignedK && gn < N && gk + 8 <= K) {
        cp16(&Bt[(long)gn * K + gk], &Bs[n * LDT + c]);
      } else {
#pragma unroll
        for (int j = 0; j < 8; j++)
          Bs[n * LDT + c + j] = (gn < N && gk + j < K) ? Bt[(long)gn * K + gk + j]
                                                       : (unsigned short)0;
      }
    }
#if HAVE_ASYNC
    asm volatile("s_wait_asynccnt 0" ::: "memory");
#endif
    __syncthreads();

    union { v16bf v; u32x4 q[2]; } af[2], bfr[4];
#pragma unroll
    for (int tr = 0; tr < 2; tr++) {
      int m = warpM + tr * 16 + laneLo;
      af[tr].q[0] = *(const u32x4*)&As[m * LDT + laneHf * 8];
      af[tr].q[1] = *(const u32x4*)&As[m * LDT + laneHf * 8 + 16];
    }
#pragma unroll
    for (int tc = 0; tc < 4; tc++) {
      int n = warpN + tc * 16 + laneLo;
      bfr[tc].q[0] = *(const u32x4*)&Bs[n * LDT + laneHf * 8];
      bfr[tc].q[1] = *(const u32x4*)&Bs[n * LDT + laneHf * 8 + 16];
    }
#pragma unroll
    for (int tr = 0; tr < 2; tr++)
#pragma unroll
      for (int tc = 0; tc < 4; tc++)
        acc[tr][tc] = __builtin_amdgcn_wmma_f32_16x16x32_bf16(
            false, af[tr].v, false, bfr[tc].v, (short)0, acc[tr][tc], false, false);
    __syncthreads();
  }

#pragma unroll
  for (int tr = 0; tr < 2; tr++)
#pragma unroll
    for (int tc = 0; tc < 4; tc++) {
      int col = colBase + warpN + tc * 16 + laneLo;
#pragma unroll
      for (int i = 0; i < 8; i++) {
        int row = rowBase + warpM + tr * 16 + laneHf * 8 + i;
        if (row < M && col < N) {
          float v = acc[tr][tc][i];
          if (bias) v += bias[row];
          C[(long)row * N + col] = v;
        }
      }
    }
}

// ---------------------------------------------------------------------------
// Fused cosine-sim GEMM + mean over points, double-buffered async pipeline:
// G[k,b] = (1/NPTS) * sum_n (Cn[KCENT,DIM] x XnT[BNCOL,DIM]^T)[k, b*NPTS+n]
// All tiles full (1024/128/32 divisible) -> pure 16B async chunk staging.
// grid = (B_, KCENT/128)
// ---------------------------------------------------------------------------
__device__ __forceinline__ void stage_pair(const unsigned short* __restrict__ gA,
                                           const unsigned short* __restrict__ gB,
                                           unsigned short* lA, unsigned short* lB,
                                           int tid)
{
#pragma unroll
  for (int h = 0; h < 2; h++) {
    int q = tid + h * 256;
    int r = q >> 2, c = (q & 3) * 8;
    cp16(gA + (long)r * DIM + c, lA + r * LDT + c);
  }
#pragma unroll
  for (int h = 0; h < 2; h++) {
    int q = tid + h * 256;
    int r = q >> 2, c = (q & 3) * 8;
    cp16(gB + (long)r * DIM + c, lB + r * LDT + c);
  }
}

__global__ __launch_bounds__(256)
void k_sim_mean(const unsigned short* __restrict__ Cn,   // [KCENT, DIM]
                const unsigned short* __restrict__ XnT,  // [BNCOL, DIM]
                float* __restrict__ G, unsigned short* __restrict__ Gbf)
{
  __shared__ __align__(16) unsigned short As[2][128 * LDT];
  __shared__ __align__(16) unsigned short Bs[2][128 * LDT];
  __shared__ float rsum[2][128];
  const int tid    = threadIdx.x;
  const int lane   = tid & 31, wid = tid >> 5;
  const int warpM  = (wid & 3) * 32;
  const int warpNi = wid >> 2;
  const int warpN  = warpNi * 64;
  const int laneLo = lane & 15, laneHf = lane >> 4;
  const int rowBase = blockIdx.y * 128;
  const int b = blockIdx.x;

  v8f acc[2][4];
  for (int tr = 0; tr < 2; tr++)
    for (int tc = 0; tc < 4; tc++)
      for (int i = 0; i < 8; i++) acc[tr][tc][i] = 0.f;

  const int T = (NPTS / 128) * (DIM / 32);  // 512 tiles
  auto bases = [&](int t, const unsigned short*& gA, const unsigned short*& gB) {
    int ct = t >> 5, k0 = (t & 31) * 32;
    gA = Cn + (long)rowBase * DIM + k0;
    gB = XnT + ((long)b * NPTS + ct * 128) * DIM + k0;
  };

  { const unsigned short *gA, *gB; bases(0, gA, gB);
    stage_pair(gA, gB, As[0], Bs[0], tid); }

  for (int t = 0; t < T; t++) {
    const int cur = t & 1;
    if (t + 1 < T) {
      const unsigned short *gA, *gB; bases(t + 1, gA, gB);
      stage_pair(gA, gB, As[cur ^ 1], Bs[cur ^ 1], tid);
    }
#if HAVE_ASYNC
    if (t + 1 < T) asm volatile("s_wait_asynccnt 4" ::: "memory");
    else           asm volatile("s_wait_asynccnt 0" ::: "memory");
#endif
    __syncthreads();

    union { v16bf v; u32x4 q[2]; } af[2], bfr[4];
#pragma unroll
    for (int tr = 0; tr < 2; tr++) {
      int m = warpM + tr * 16 + laneLo;
      af[tr].q[0] = *(const u32x4*)&As[cur][m * LDT + laneHf * 8];
      af[tr].q[1] = *(const u32x4*)&As[cur][m * LDT + laneHf * 8 + 16];
    }
#pragma unroll
    for (int tc = 0; tc < 4; tc++) {
      int n = warpN + tc * 16 + laneLo;
      bfr[tc].q[0] = *(const u32x4*)&Bs[cur][n * LDT + laneHf * 8];
      bfr[tc].q[1] = *(const u32x4*)&Bs[cur][n * LDT + laneHf * 8 + 16];
    }
#pragma unroll
    for (int tr = 0; tr < 2; tr++)
#pragma unroll
      for (int tc = 0; tc < 4; tc++)
        acc[tr][tc] = __builtin_amdgcn_wmma_f32_16x16x32_bf16(
            false, af[tr].v, false, bfr[tc].v, (short)0, acc[tr][tc], false, false);
    __syncthreads();
  }

  // reduce the 64 owned columns per wave -> per-row sums -> mean
#pragma unroll
  for (int tr = 0; tr < 2; tr++) {
    float p[8];
#pragma unroll
    for (int i = 0; i < 8; i++)
      p[i] = acc[tr][0][i] + acc[tr][1][i] + acc[tr][2][i] + acc[tr][3][i];
#pragma unroll
    for (int m = 1; m < 16; m <<= 1)
#pragma unroll
      for (int i = 0; i < 8; i++) p[i] += __shfl_xor(p[i], m, 32);
    if (laneLo == 0) {
      int rb = warpM + tr * 16 + laneHf * 8;
#pragma unroll
      for (int i = 0; i < 8; i++) rsum[warpNi][rb + i] = p[i];
    }
  }
  __syncthreads();
  if (tid < 128) {
    float v = (rsum[0][tid] + rsum[1][tid]) * (1.0f / NPTS);
    int row = rowBase + tid;
    G[row * B_ + b]            = v;       // f32, channel-major (debug/unused)
    Gbf[(long)b * DIM + row]   = f2bf(v); // transposed [B, DIM] for fc1 B-op
  }
}

// ---------------------------------------------------------------------------
// Utility / elementwise kernels
// ---------------------------------------------------------------------------
__global__ void k_transpose_in(const float* __restrict__ in,
                               float* __restrict__ Xcm,          // [3, BNCOL] f32
                               unsigned short* __restrict__ Xbf) // [BNCOL, 3] bf16
{
  int col = blockIdx.x * blockDim.x + threadIdx.x;
  if (col >= BNCOL) return;
  int b = col / NPTS, n = col % NPTS;
#pragma unroll
  for (int c = 0; c < 3; c++) {
    float v = in[((long)b * 3 + c) * NPTS + n];
    Xcm[(long)c * BNCOL + col] = v;
    Xbf[(long)col * 3 + c] = f2bf(v);
  }
}

__global__ void k_conv_bf(const float* __restrict__ src, unsigned short* __restrict__ dst,
                          int rows, int cols, int srcLd, int colOff)
{
  long total = (long)rows * cols;
  for (long i = (long)blockIdx.x * blockDim.x + threadIdx.x;
       i < total; i += (long)gridDim.x * blockDim.x) {
    int r = (int)(i / cols), c = (int)(i % cols);
    dst[i] = f2bf(src[(long)r * srcLd + colOff + c]);
  }
}

__global__ void k_row_stats(const float* __restrict__ X, int L,
                            float* __restrict__ mean, float* __restrict__ var)
{
  int c = blockIdx.x;
  const float* row = X + (long)c * L;
  float s = 0.f, s2 = 0.f;
  for (int i = threadIdx.x; i < L; i += blockDim.x) { float v = row[i]; s += v; s2 += v * v; }
  for (int m = 1; m < 32; m <<= 1) { s += __shfl_xor(s, m, 32); s2 += __shfl_xor(s2, m, 32); }
  __shared__ float sh[2][8];
  int wid = threadIdx.x >> 5;
  if ((threadIdx.x & 31) == 0) { sh[0][wid] = s; sh[1][wid] = s2; }
  __syncthreads();
  if (threadIdx.x == 0) {
    float S = 0.f, S2 = 0.f;
    for (int w = 0; w < 8; w++) { S += sh[0][w]; S2 += sh[1][w]; }
    float m_ = S / L;
    mean[c] = m_; var[c] = S2 / L - m_ * m_;
  }
}

// y = g*(x-mean)*rsqrt(var+eps)+b (+ReLU); bf16 copy TRANSPOSED [L, C]
__global__ void k_bn_act(const float* __restrict__ X, float* __restrict__ Y,
                         unsigned short* __restrict__ Ybf,
                         const float* __restrict__ mean, const float* __restrict__ var,
                         const float* __restrict__ g, const float* __restrict__ b,
                         int C, long L, int relu)
{
  long total = (long)C * L;
  for (long i = (long)blockIdx.x * blockDim.x + threadIdx.x;
       i < total; i += (long)gridDim.x * blockDim.x) {
    int c = (int)(i / L);
    long col = i - (long)c * L;
    float sc = g[c] * rsqrtf(var[c] + 1e-5f);
    float sh = b[c] - mean[c] * sc;
    float v = X[i] * sc + sh;
    if (relu) v = fmaxf(v, 0.f);
    Y[i] = v;
    if (Ybf) Ybf[col * C + c] = f2bf(v);   // transposed [L, C] for GEMM B-op
  }
}

__global__ __launch_bounds__(256)
void k_colmax(const float* __restrict__ X, float* __restrict__ G,
              unsigned short* __restrict__ Gbf)
{
  int c = blockIdx.x >> 5;   // / B_
  int b = blockIdx.x & 31;
  const float* p = X + (long)c * BNCOL + (long)b * NPTS;
  float m = -3.402823466e38f;
  for (int i = threadIdx.x; i < NPTS; i += 256) m = fmaxf(m, p[i]);
  for (int s = 1; s < 32; s <<= 1) m = fmaxf(m, __shfl_xor(m, s, 32));
  __shared__ float sh[8];
  if ((threadIdx.x & 31) == 0) sh[threadIdx.x >> 5] = m;
  __syncthreads();
  if (threadIdx.x == 0) {
    float mm = sh[0];
    for (int w = 1; w < 8; w++) mm = fmaxf(mm, sh[w]);
    G[c * B_ + b] = mm;
    Gbf[(long)b * 1024 + c] = f2bf(mm);    // transposed [B, 1024]
  }
}

__global__ void k_make_trans(const float* __restrict__ F /*[k2,B]*/,
                             float* __restrict__ T /*[B,k2]*/, int k)
{
  int k2 = k * k;
  long total = (long)B_ * k2;
  long i = (long)blockIdx.x * blockDim.x + threadIdx.x;
  if (i >= total) return;
  int b = (int)(i / k2), r = (int)(i % k2);
  T[i] = F[(long)r * B_ + b] + ((r % (k + 1)) == 0 ? 1.f : 0.f);
}

__global__ void k_apply_trans3(const float* __restrict__ Xcm, const float* __restrict__ T,
                               unsigned short* __restrict__ Xout /*[BNCOL,3]*/)
{
  int col = blockIdx.x * blockDim.x + threadIdx.x;
  if (col >= BNCOL) return;
  int b = col / NPTS;
  const float* t = T + b * 9;
  float x0 = Xcm[col], x1 = Xcm[BNCOL + col], x2 = Xcm[2 * BNCOL + col];
#pragma unroll
  for (int k = 0; k < 3; k++) {
    float v = x0 * t[0 * 3 + k] + x1 * t[1 * 3 + k] + x2 * t[2 * 3 + k];
    Xout[(long)col * 3 + k] = f2bf(v);
  }
}

__global__ __launch_bounds__(256)
void k_apply_trans64(const float* __restrict__ H, const float* __restrict__ T,
                     unsigned short* __restrict__ Xout /*[BNCOL,64]*/)
{
  __shared__ float Ts[4096];
  int col = blockIdx.x * 256 + threadIdx.x;   // 256 divides NPTS -> one batch per block
  int b = col / NPTS;
  const float* t = T + (long)b * 4096;
  for (int i = threadIdx.x; i < 4096; i += 256) Ts[i] = t[i];
  __syncthreads();
  float h[64];
#pragma unroll
  for (int c = 0; c < 64; c++) h[c] = H[(long)c * BNCOL + col];
  for (int k = 0; k < 64; k++) {
    float a = 0.f;
#pragma unroll
    for (int c = 0; c < 64; c++) a += h[c] * Ts[c * 64 + k];
    Xout[(long)col * 64 + k] = f2bf(a);
  }
}

// per-point inverse L2 norm (coalesced reads of [DIM, BNCOL])
__global__ void k_colnorm(const float* __restrict__ X, float* __restrict__ invn)
{
  int col = blockIdx.x * blockDim.x + threadIdx.x;
  if (col >= BNCOL) return;
  float s = 0.f;
  for (int r = 0; r < DIM; r++) { float v = X[(long)r * BNCOL + col]; s += v * v; }
  invn[col] = 1.f / (sqrtf(s) + 1e-8f);
}

// LDS-tiled transpose+scale: XnT[col, r] = X[r, col] * invn[col], bf16
__global__ __launch_bounds__(256)
void k_tr_scale(const float* __restrict__ X, const float* __restrict__ invn,
                unsigned short* __restrict__ XnT)
{
  __shared__ float tile[64][65];
  int r0 = blockIdx.y * 64, c0 = blockIdx.x * 64;
#pragma unroll
  for (int j = 0; j < 16; j++) {
    int r = (threadIdx.x >> 6) + j * 4;
    int c = threadIdx.x & 63;
    tile[r][c] = X[(long)(r0 + r) * BNCOL + c0 + c];
  }
  __syncthreads();
#pragma unroll
  for (int j = 0; j < 16; j++) {
    int c = (threadIdx.x >> 6) + j * 4;   // point
    int r = threadIdx.x & 63;             // channel (contiguous in output)
    XnT[(long)(c0 + c) * DIM + r0 + r] = f2bf(tile[r][c] * invn[c0 + c]);
  }
}

__global__ __launch_bounds__(256)
void k_norm_rows(const float* __restrict__ C, unsigned short* __restrict__ Cn)
{
  int k = blockIdx.x;
  const float* row = C + (long)k * DIM;
  float s = 0.f;
  for (int i = threadIdx.x; i < DIM; i += 256) { float v = row[i]; s += v * v; }
  for (int m = 1; m < 32; m <<= 1) s += __shfl_xor(s, m, 32);
  __shared__ float sh[8];
  __shared__ float inv;
  if ((threadIdx.x & 31) == 0) sh[threadIdx.x >> 5] = s;
  __syncthreads();
  if (threadIdx.x == 0) {
    float S = 0.f;
    for (int w = 0; w < 8; w++) S += sh[w];
    inv = 1.f / (sqrtf(S) + 1e-8f);
  }
  __syncthreads();
  float iv = inv;
  for (int i = threadIdx.x; i < DIM; i += 256) Cn[(long)k * DIM + i] = f2bf(row[i] * iv);
}

// R[pr, o] = leaky(U[o,n] + V[o,b] + b1[o]); pr = b*NATTR+n ; bf16 [NPAIR, ADIM]
__global__ void k_pairs(const float* __restrict__ U, const float* __restrict__ V,
                        const float* __restrict__ b1, unsigned short* __restrict__ R)
{
  long total = (long)NPAIR * ADIM;
  for (long i = (long)blockIdx.x * blockDim.x + threadIdx.x;
       i < total; i += (long)gridDim.x * blockDim.x) {
    int pr = (int)(i / ADIM), o = (int)(i % ADIM);
    int b = pr / NATTR, n = pr % NATTR;
    float v = U[o * NATTR + n] + V[o * B_ + b] + b1[o];
    v = v > 0.f ? v : 0.01f * v;
    R[i] = f2bf(v);
  }
}

__global__ void k_pred(const float* __restrict__ S, const float* __restrict__ w3,
                       const float* __restrict__ b3, float* __restrict__ pred)
{
  int pr = blockIdx.x * blockDim.x + threadIdx.x;
  if (pr >= NPAIR) return;
  float a = 0.f;
  for (int p = 0; p < 600; p++) {
    float v = S[(long)p * NPAIR + pr];
    v = v > 0.f ? v : 0.01f * v;
    a += w3[p] * v;
  }
  a += b3[0];
  pred[pr] = 1.f / (1.f + expf(-a));
}

__global__ void k_feats_out(const float* __restrict__ Fc, float* __restrict__ out)
{
  int i = blockIdx.x * blockDim.x + threadIdx.x;
  if (i >= B_ * ADIM) return;
  int b = i / ADIM, o = i % ADIM;
  out[i] = Fc[o * B_ + b];
}

// ---------------------------------------------------------------------------
// Host orchestration
// ---------------------------------------------------------------------------
extern "C" void kernel_launch(void* const* d_in, const int* in_sizes, int n_in,
                              void* d_out, int out_size, void* d_ws, size_t ws_size,
                              hipStream_t stream)
{
  (void)in_sizes; (void)n_in; (void)out_size; (void)ws_size;

  const float* input     = (const float*)d_in[0];
  const float* attribute = (const float*)d_in[1];
  void* const* stnP  = d_in + 2;    // 22 entries: c1w..f3b, bn1g..bn5b
  void* const* fstnP = d_in + 24;   // 22 entries
  const float* c1w  = (const float*)d_in[46]; const float* c1b  = (const float*)d_in[47];
  const float* c2w  = (const float*)d_in[48]; const float* c2b  = (const float*)d_in[49];
  const float* c3w  = (const float*)d_in[50]; const float* c3b  = (const float*)d_in[51];
  const float* bn1g = (const float*)d_in[52]; const float* bn1b = (const float*)d_in[53];
  const float* bn2g = (const float*)d_in[54]; const float* bn2b = (const float*)d_in[55];
  const float* bn3g = (const float*)d_in[56]; const float* bn3b = (const float*)d_in[57];
  const float* centroids = (const float*)d_in[58];
  const float* fc1w = (const float*)d_in[59]; const float* fc1b = (const float*)d_in[60];
  const float* fbn1g= (const float*)d_in[61]; const float* fbn1b= (const float*)d_in[62];
  const float* fc2w = (const float*)d_in[63]; const float* fc2b = (const float*)d_in[64];
  const float* fbn2g= (const float*)d_in[65]; const float* fbn2b= (const float*)d_in[66];
  const float* rn1w = (const float*)d_in[67]; const float* rn1b = (const float*)d_in[68];
  const float* rn2w = (const float*)d_in[69]; const float* rn2b = (const float*)d_in[70];
  const float* rn3w = (const float*)d_in[71]; const float* rn3b = (const float*)d_in[72];

  float* outFeats = (float*)d_out;                  // [32,300]
  float* outTF    = outFeats + B_ * ADIM;           // [32,64,64]
  float* outPred  = outTF + B_ * 64 * 64;           // [32,100]

  size_t off = 0;
  auto alloc = [&](size_t bytes) -> void* {
    off = (off + 255) & ~(size_t)255;
    void* p = (char*)d_ws + off;
    off += bytes;
    return p;
  };
  auto cdiv = [](long a, long b) -> int { return (int)((a + b - 1) / b); };

  float*          X3f   = (float*)alloc(3L * BNCOL * 4);          // [3, BNCOL]
  unsigned short* X3h   = (unsigned short*)alloc(3L * BNCOL * 2); // [BNCOL, 3]
  unsigned short* X3th  = (unsigned short*)alloc(3L * BNCOL * 2); // [BNCOL, 3]
  float*          P64a  = (float*)alloc(64L * BNCOL * 4);
  unsigned short* Q64a  = (unsigned short*)alloc(64L * BNCOL * 2); // [BNCOL, 64]
  float*          P64b  = (float*)alloc(64L * BNCOL * 4);
  unsigned short* Q64b  = (unsigned short*)alloc(64L * BNCOL * 2); // [BNCOL, 64]
  float*          P128  = (float*)alloc(128L * BNCOL * 4);
  unsigned short* Q128  = (unsigned short*)alloc(128L * BNCOL * 2); // [BNCOL, 128]
  float*          P1024 = (float*)alloc(1024L * BNCOL * 4);
  unsigned short* Q1024 = (unsigned short*)alloc(1024L * BNCOL * 2); // XnT [BNCOL, DIM]
  unsigned short* Cnh   = (unsigned short*)alloc((long)KCENT * DIM * 2);
  float*          invn  = (float*)alloc((long)BNCOL * 4);
  float*          Gf    = (float*)alloc(1024L * B_ * 4);
  unsigned short* Gh    = (unsigned short*)alloc(1024L * B_ * 2);  // [B, 1024]
  float*          F1f   = (float*)alloc(512L * B_ * 4);
  unsigned short* F1h   = (unsigned short*)alloc(512L * B_ * 2);   // [B, 512]
  float*          F2f   = (float*)alloc(256L * B_ * 4);
  unsigned short* F2h   = (unsigned short*)alloc(256L * B_ * 2);   // [B, 256]
  float*          F9    = (float*)alloc(9L * B_ * 4);
  float*          F4096 = (float*)alloc(4096L * B_ * 4);
  float*          T9    = (float*)alloc((long)B_ * 9 * 4);
  float*          FeatF = (float*)alloc((long)ADIM * B_ * 4);
  unsigned short* FeatH = (unsigned short*)alloc((long)ADIM * B_ * 2); // [B, 300]
  float*          Uf    = (float*)alloc((long)ADIM * NATTR * 4);
  float*          Vf    = (float*)alloc((long)ADIM * B_ * 4);
  unsigned short* Rh    = (unsigned short*)alloc((long)ADIM * NPAIR * 2); // [NPAIR, 300]
  float*          Sf    = (float*)alloc(600L * NPAIR * 4);
  unsigned short* attrTh= (unsigned short*)alloc((long)ADIM * NATTR * 2); // [100, 300]
  unsigned short* rn1Ah = (unsigned short*)alloc(300L * 300 * 2);
  unsigned short* rn1Fh = (unsigned short*)alloc(300L * 300 * 2);
  float*          meanS = (float*)alloc(1024 * 4);
  float*          varS  = (float*)alloc(1024 * 4);

  auto toBF = [&](const float* w, int rows, int cols) -> unsigned short* {
    unsigned short* dst = (unsigned short*)alloc((size_t)rows * cols * 2);
    int g = cdiv((long)rows * cols, 256); if (g > 4096) g = 4096;
    k_conv_bf<<<g, 256, 0, stream>>>(w, dst, rows, cols, cols, 0);
    return dst;
  };
  auto gemm = [&](const unsigned short* A, const unsigned short* Bt, float* C,
                  const float* bias, int M, int N, int K) {
    dim3 g(cdiv(N, 128), cdiv(M, 128));
    k_gemm<<<g, 256, 0, stream>>>(A, Bt, C, bias, M, N, K);
  };
  auto bnact = [&](float* X, int C, long L, const float* g_, const float* b_,
                   unsigned short* Ybf, int relu) {
    k_row_stats<<<C, 256, 0, stream>>>(X, (int)L, meanS, varS);
    int g = cdiv((long)C * L, 256); if (g > 4096) g = 4096;
    k_bn_act<<<g, 256, 0, stream>>>(X, X, Ybf, meanS, varS, g_, b_, C, L, relu);
  };

  auto runSTN = [&](void* const* P, const unsigned short* Xbf, int Cin, int k,
                    float* A64f, unsigned short* A64h, float* Fout) {
    unsigned short* w;
    w = toBF((const float*)P[0], 64, Cin);
    gemm(w, Xbf, A64f, (const float*)P[1], 64, BNCOL, Cin);
    bnact(A64f, 64, BNCOL, (const float*)P[12], (const float*)P[13], A64h, 1);
    w = toBF((const float*)P[2], 128, 64);
    gemm(w, A64h, P128, (const float*)P[3], 128, BNCOL, 64);
    bnact(P128, 128, BNCOL, (const float*)P[14], (const float*)P[15], Q128, 1);
    w = toBF((const float*)P[4], 1024, 128);
    gemm(w, Q128, P1024, (const float*)P[5], 1024, BNCOL, 128);
    bnact(P1024, 1024, BNCOL, (const float*)P[16], (const float*)P[17], nullptr, 1);
    k_colmax<<<1024 * B_, 256, 0, stream>>>(P1024, Gf, Gh);
    w = toBF((const float*)P[6], 512, 1024);
    gemm(w, Gh, F1f, (const float*)P[7], 512, B_, 1024);
    bnact(F1f, 512, B_, (const float*)P[18], (const float*)P[19], F1h, 1);
    w = toBF((const float*)P[8], 256, 512);
    gemm(w, F1h, F2f, (const float*)P[9], 256, B_, 512);
    bnact(F2f, 256, B_, (const float*)P[20], (const float*)P[21], F2h, 1);
    w = toBF((const float*)P[10], k * k, 256);
    gemm(w, F2h, Fout, (const float*)P[11], k * k, B_, 256);
  };

  unsigned short* w;

  // 1) input [B,3,N] -> f32 [3,BNCOL] + bf16 [BNCOL,3]
  k_transpose_in<<<cdiv(BNCOL, 256), 256, 0, stream>>>(input, X3f, X3h);
  // 2) STN3d -> F9 [9, B]
  runSTN(stnP, X3h, 3, 3, P64a, Q64a, F9);
  // 3) trans = F9^T + I  -> T9 [B,9]
  k_make_trans<<<cdiv((long)B_ * 9, 256), 256, 0, stream>>>(F9, T9, 3);
  // 4) apply 3x3 transform -> X3th [BNCOL,3] bf16
  k_apply_trans3<<<cdiv(BNCOL, 256), 256, 0, stream>>>(X3f, T9, X3th);
  // 5) conv1 3->64 + BN + ReLU
  w = toBF(c1w, 64, 3);
  gemm(w, X3th, P64a, c1b, 64, BNCOL, 3);
  bnact(P64a, 64, BNCOL, bn1g, bn1b, Q64a, 1);
  // 6) STNkd(64) -> F4096 [4096, B]
  runSTN(fstnP, Q64a, 64, 64, P64b, Q64b, F4096);
  // 7) trans_feat = F4096^T + I -> d_out [B,64,64]
  k_make_trans<<<cdiv((long)B_ * 4096, 256), 256, 0, stream>>>(F4096, outTF, 64);
  // 8) apply 64x64 transform -> Q64b [BNCOL,64] bf16
  k_apply_trans64<<<BNCOL / 256, 256, 0, stream>>>(P64a, outTF, Q64b);
  // 9) conv2 64->128 + BN + ReLU
  w = toBF(c2w, 128, 64);
  gemm(w, Q64b, P128, c2b, 128, BNCOL, 64);
  bnact(P128, 128, BNCOL, bn2g, bn2b, Q128, 1);
  // 10) conv3 128->1024 + BN (no ReLU)
  w = toBF(c3w, 1024, 128);
  gemm(w, Q128, P1024, c3b, 1024, BNCOL, 128);
  bnact(P1024, 1024, BNCOL, bn3g, bn3b, nullptr, 0);
  // 11) per-point norms + tiled transpose/scale -> XnT [BNCOL, DIM] bf16
  k_colnorm<<<cdiv(BNCOL, 256), 256, 0, stream>>>(P1024, invn);
  { dim3 g(BNCOL / 64, DIM / 64); k_tr_scale<<<g, 256, 0, stream>>>(P1024, invn, Q1024); }
  // 12) row-normalize centroids -> Cnh [KCENT, DIM] bf16
  k_norm_rows<<<KCENT, 256, 0, stream>>>(centroids, Cnh);
  // 13) fused cosine-sim GEMM + mean (double-buffered async) -> Gf/Gh
  { dim3 g(B_, KCENT / 128); k_sim_mean<<<g, 256, 0, stream>>>(Cnh, Q1024, Gf, Gh); }
  // 14) fc1 1024->512 + BN + ReLU
  w = toBF(fc1w, 512, 1024);
  gemm(w, Gh, F1f, fc1b, 512, B_, 1024);
  bnact(F1f, 512, B_, fbn1g, fbn1b, F1h, 1);
  // 15) fc2 512->300 + BN + ReLU -> feats
  w = toBF(fc2w, 300, 512);
  gemm(w, F1h, FeatF, fc2b, 300, B_, 512);
  bnact(FeatF, 300, B_, fbn2g, fbn2b, FeatH, 1);
  // 16) feats -> d_out [B,300]
  k_feats_out<<<cdiv((long)B_ * ADIM, 256), 256, 0, stream>>>(FeatF, outFeats);
  // 17) attribute -> bf16 [100,300] (already [N,K] for the U GEMM)
  k_conv_bf<<<cdiv((long)NATTR * ADIM, 256), 256, 0, stream>>>(attribute, attrTh, NATTR, ADIM, ADIM, 0);
  // 18) rn1 weight split (attr half / feat half) bf16
  k_conv_bf<<<cdiv(300L * 300, 256), 256, 0, stream>>>(rn1w, rn1Ah, 300, 300, 600, 0);
  k_conv_bf<<<cdiv(300L * 300, 256), 256, 0, stream>>>(rn1w, rn1Fh, 300, 300, 600, 300);
  // 19/20) U = rn1A x attr^T [300,100];  V = rn1F x feats^T [300,32]
  gemm(rn1Ah, attrTh, Uf, nullptr, 300, NATTR, 300);
  gemm(rn1Fh, FeatH, Vf, nullptr, 300, B_, 300);
  // 21) leaky(U + V + b1) -> Rh [NPAIR, 300] bf16
  { int g = cdiv((long)NPAIR * ADIM, 256); if (g > 4096) g = 4096;
    k_pairs<<<g, 256, 0, stream>>>(Uf, Vf, rn1b, Rh); }
  // 22) rn2 GEMM [600,300] x [NPAIR,300]^T -> S [600, NPAIR]
  w = toBF(rn2w, 600, 300);
  gemm(w, Rh, Sf, rn2b, 600, NPAIR, 300);
  // 23) pred = sigmoid(rn3 . leaky(S)) -> d_out [B,100]
  k_pred<<<cdiv(NPAIR, 256), 256, 0, stream>>>(Sf, rn3w, rn3b, outPred);
}
